// AltAttention_29609504539092
// MI455X (gfx1250) — matrix operand order repaired and checked by reading, not compiled, not run on hardware
//
#include <hip/hip_runtime.h>

typedef __attribute__((ext_vector_type(16))) _Float16 v16h;
typedef __attribute__((ext_vector_type(8)))  float    v8f;

constexpr int NH = 8;        // heads
constexpr int DM = 512;      // model dim
constexpr int DHd = 64;      // head dim
constexpr int SQ = 2048;     // seq len
constexpr int NB = 4;        // batch
constexpr float SCL = 0.044194173824159216f;  // 512^-0.5 (note: DIM, not Dh)

// K-index inside a 16x32 f16 A-fragment for element e of lane-half hh
__device__ __forceinline__ int a_elem_k(int e, int hh) {
  return e + 16 * hh;
}

// ---------------------------------------------------------------------------
// Elementwise cast: hidden fp32 -> f16
// ---------------------------------------------------------------------------
__global__ __launch_bounds__(256) void hcast_kernel(const float* __restrict__ X,
                                                    _Float16* __restrict__ Xh,
                                                    int n) {
  int idx = blockIdx.x * 256 + threadIdx.x;
  if (idx < n) Xh[idx] = (_Float16)X[idx];
}

// ---------------------------------------------------------------------------
// Weight prep: Wt[n][k] = (f16) W[k][n]
// ---------------------------------------------------------------------------
__global__ __launch_bounds__(256) void wcast_kernel(const float* __restrict__ W,
                                                    _Float16* __restrict__ Wt,
                                                    int Kdim, int Ndim) {
  int idx = blockIdx.x * 256 + threadIdx.x;
  if (idx >= Kdim * Ndim) return;
  int n = idx / Kdim;
  int k = idx - n * Kdim;
  Wt[idx] = (_Float16)W[(size_t)k * Ndim + n];
}

// ---------------------------------------------------------------------------
// QKV projection: Xh[B*S,512](f16) @ Wqkv -> Q (scaled, [bh][s][d]) /
// K ([bh][s][d]) / V transposed ([bh][d][s]).
// A-tile double-buffered in LDS via async global->LDS copies.
// ---------------------------------------------------------------------------
__global__ __launch_bounds__(128) void qkv_kernel(
    const _Float16* __restrict__ Xh, const _Float16* __restrict__ Wt,
    const float* __restrict__ bias,
    _Float16* __restrict__ Qh, _Float16* __restrict__ Kh,
    _Float16* __restrict__ Vt) {
  __shared__ _Float16 As[2][16][32];  // 2 x 1KB
  int tid = threadIdx.x;
  int wave = tid >> 5, lane = tid & 31;
  int nl = lane & 15, hh = lane >> 4;
  int m0 = blockIdx.x * 16;
  int n0 = blockIdx.y * 64 + wave * 16;
  unsigned ldsA = (unsigned)(uintptr_t)&As[0][0][0];
  const _Float16* Xbase = Xh + (size_t)m0 * DM;

  auto issueA = [&](int k0, int buf) {
    if (tid < 64) {  // waves 0,1: one async op each, 64B per A-row
      int row = tid >> 2, part = tid & 3;
      unsigned long long g =
          (unsigned long long)(uintptr_t)(Xbase + (size_t)row * DM + k0) +
          (unsigned long long)part * 16u;
      asm volatile("global_load_async_to_lds_b128 %0, %1, off"
                   :: "v"(ldsA + (unsigned)buf * 1024u + (unsigned)tid * 16u),
                      "v"(g) : "memory");
    }
  };

  v8f acc = {};
  issueA(0, 0);
  for (int k0 = 0; k0 < DM; k0 += 32) {
    int cur = (k0 >> 5) & 1;
    if (k0 + 32 < DM) {
      issueA(k0 + 32, cur ^ 1);
      asm volatile("s_wait_asynccnt 0x1" ::: "memory");  // prev tile done
    } else {
      asm volatile("s_wait_asynccnt 0x0" ::: "memory");
    }
    __syncthreads();
    v16h a, bfr;
#pragma unroll
    for (int e = 0; e < 16; ++e) a[e] = As[cur][nl][a_elem_k(e, hh)];
    const _Float16* wp = &Wt[(size_t)(n0 + nl) * DM + k0 + 16 * hh];
#pragma unroll
    for (int e = 0; e < 16; ++e) bfr[e] = wp[e];
    acc = __builtin_amdgcn_wmma_f32_16x16x32_f16(false, a, false, bfr,
                                                 (short)0, acc, false, false);
    __syncthreads();  // reads of As[cur] done before it is refilled
  }
  int col = n0 + nl;
  int hd = col / 192, rem = col % 192;
  int typ = rem / 64, d = rem - typ * 64;
  float bb = bias[col];
#pragma unroll
  for (int r = 0; r < 8; ++r) {
    int m = m0 + r + 8 * hh;
    int bidx = m >> 11, s = m & 2047;
    float val = acc[r] + bb;
    size_t bh = (size_t)bidx * NH + hd;
    if (typ == 0)
      Qh[(bh * SQ + s) * DHd + d] = (_Float16)(val * SCL);
    else if (typ == 1)
      Kh[(bh * SQ + s) * DHd + d] = (_Float16)val;
    else
      Vt[(bh * DHd + d) * SQ + s] = (_Float16)val;
  }
}

// ---------------------------------------------------------------------------
// Flash attention: 4 waves/block = 4 q-tiles of one (b,h).  K/V tiles are
// double-buffered in LDS via async global->LDS copies (ASYNCcnt ping-pong).
// ---------------------------------------------------------------------------
__global__ __launch_bounds__(128) void attn_kernel(
    const _Float16* __restrict__ Qh, const _Float16* __restrict__ Kh,
    const _Float16* __restrict__ Vt, const float* __restrict__ alibi,
    const unsigned char* __restrict__ mask, _Float16* __restrict__ Ctx) {
  __shared__ _Float16 Kt[2][32 * 64];   // [key][d]   2 x 4KB
  __shared__ _Float16 Vs[2][64 * 32];   // [d][key]   2 x 4KB
  __shared__ _Float16 Ps[4][16][32];    // per-wave P 4KB

  int tid = threadIdx.x;
  int wave = tid >> 5, lane = tid & 31;
  int nl = lane & 15, hh = lane >> 4;
  int bh = blockIdx.x >> 5;             // 32 blocks of 64 q-rows per (b,h)
  int qblk = blockIdx.x & 31;
  int b = bh >> 3, h = bh & 7;
  int q0 = qblk * 64 + wave * 16;

  const _Float16* Kbase = Kh + (size_t)bh * SQ * DHd;
  const _Float16* Vbase = Vt + (size_t)bh * DHd * SQ;
  unsigned ldsK = (unsigned)(uintptr_t)&Kt[0][0];
  unsigned ldsV = (unsigned)(uintptr_t)&Vs[0][0];

  auto issueKV = [&](int j0, int buf) {
    // K tile: 32 keys x 64 d, contiguous 4KB
#pragma unroll
    for (int i = 0; i < 2; ++i) {
      int seg = tid + i * 128;  // 0..255, 16B each
      unsigned long long g =
          (unsigned long long)(uintptr_t)(Kbase + (size_t)j0 * DHd) +
          (unsigned long long)seg * 16u;
      asm volatile("global_load_async_to_lds_b128 %0, %1, off"
                   :: "v"(ldsK + (unsigned)buf * 4096u + (unsigned)seg * 16u),
                      "v"(g) : "memory");
    }
    // V tile: 64 d-rows x 32 keys, 64B per strided row
#pragma unroll
    for (int i = 0; i < 2; ++i) {
      int seg = tid + i * 128;
      int row = seg >> 2, part = seg & 3;
      unsigned long long g =
          (unsigned long long)(uintptr_t)(Vbase + (size_t)row * SQ + j0) +
          (unsigned long long)part * 16u;
      asm volatile("global_load_async_to_lds_b128 %0, %1, off"
                   :: "v"(ldsV + (unsigned)buf * 4096u + (unsigned)seg * 16u),
                      "v"(g) : "memory");
    }
  };

  // Q fragments (16 x 64 = two 16x32 A-fragments), kept in registers
  v16h qa[2];
#pragma unroll
  for (int kk = 0; kk < 2; ++kk)
#pragma unroll
    for (int e = 0; e < 16; ++e)
      qa[kk][e] =
          Qh[((size_t)bh * SQ + q0 + nl) * DHd + 32 * kk + a_elem_k(e, hh)];

  v8f o0 = {}, o1 = {}, o2 = {}, o3 = {};
  float mrun[8], lrun[8];
#pragma unroll
  for (int r = 0; r < 8; ++r) { mrun[r] = -1e30f; lrun[r] = 0.f; }

  const float* ab = alibi + (size_t)h * SQ * SQ;

  issueKV(0, 0);
  for (int j0 = 0; j0 < SQ; j0 += 32) {
    int cur = (j0 >> 5) & 1;
    if (j0 + 32 < SQ) {
      issueKV(j0 + 32, cur ^ 1);
      asm volatile("s_wait_asynccnt 0x4" ::: "memory");  // prev tile's 4 done
    } else {
      asm volatile("s_wait_asynccnt 0x0" ::: "memory");
    }
    __syncthreads();  // all waves' tiles visible

    v8f st[2];
#pragma unroll
    for (int t = 0; t < 2; ++t) {
      int key = j0 + 16 * t + nl;
      v8f s = {};
#pragma unroll
      for (int kk = 0; kk < 2; ++kk) {
        v16h bf;
        const _Float16* kp = &Kt[cur][(16 * t + nl) * 64 + 32 * kk + 16 * hh];
#pragma unroll
        for (int e = 0; e < 16; ++e) bf[e] = kp[e];
        s = __builtin_amdgcn_wmma_f32_16x16x32_f16(false, qa[kk], false, bf,
                                                   (short)0, s, false, false);
      }
      bool mk = mask[b * SQ + key] != 0;
#pragma unroll
      for (int r = 0; r < 8; ++r) {
        float v = s[r] + ab[(size_t)(q0 + r + 8 * hh) * SQ + key];
        s[r] = mk ? v : -1e30f;
      }
      st[t] = s;
    }
    // online softmax row statistics (rows r + 8*hh, reduced over 16 lanes)
#pragma unroll
    for (int r = 0; r < 8; ++r) {
      float pm = fmaxf(st[0][r], st[1][r]);
#pragma unroll
      for (int sm = 1; sm < 16; sm <<= 1) pm = fmaxf(pm, __shfl_xor(pm, sm, 32));
      float newm = fmaxf(mrun[r], pm);
      float alpha = __expf(mrun[r] - newm);
      mrun[r] = newm;
      float p0 = __expf(st[0][r] - newm);
      float p1 = __expf(st[1][r] - newm);
      st[0][r] = p0; st[1][r] = p1;
      float ps = p0 + p1;
#pragma unroll
      for (int sm = 1; sm < 16; sm <<= 1) ps += __shfl_xor(ps, sm, 32);
      lrun[r] = lrun[r] * alpha + ps;
      o0[r] *= alpha; o1[r] *= alpha; o2[r] *= alpha; o3[r] *= alpha;
    }
    // transpose P (C layout -> A layout) via per-wave LDS buffer
#pragma unroll
    for (int t = 0; t < 2; ++t)
#pragma unroll
      for (int r = 0; r < 8; ++r)
        Ps[wave][r + 8 * hh][16 * t + nl] = (_Float16)st[t][r];
    __syncthreads();
    v16h pa;
#pragma unroll
    for (int e = 0; e < 16; ++e) pa[e] = Ps[wave][nl][a_elem_k(e, hh)];
    // O += P @ V  (four d-tiles of 16)
#pragma unroll
    for (int dt = 0; dt < 4; ++dt) {
      v16h bf;
      const _Float16* vp = &Vs[cur][(16 * dt + nl) * 32 + 16 * hh];
#pragma unroll
      for (int e = 0; e < 16; ++e) bf[e] = vp[e];
      v8f* op = (dt == 0) ? &o0 : (dt == 1) ? &o1 : (dt == 2) ? &o2 : &o3;
      *op = __builtin_amdgcn_wmma_f32_16x16x32_f16(false, pa, false, bf,
                                                   (short)0, *op, false, false);
    }
    __syncthreads();  // reads of Kt/Vs[cur] done before it is refilled
  }
#pragma unroll
  for (int r = 0; r < 8; ++r) {
    float inv = 1.0f / lrun[r];
    size_t row = ((size_t)b * SQ + q0 + r + 8 * hh) * DM + (size_t)h * DHd;
    Ctx[row + 0  + nl] = (_Float16)(o0[r] * inv);
    Ctx[row + 16 + nl] = (_Float16)(o1[r] * inv);
    Ctx[row + 32 + nl] = (_Float16)(o2[r] * inv);
    Ctx[row + 48 + nl] = (_Float16)(o3[r] * inv);
  }
}

// ---------------------------------------------------------------------------
// Output projection: Ctx[B*S,512](f16) @ Wp + b -> fp32 out.
// A-tile double-buffered in LDS via async global->LDS copies.
// ---------------------------------------------------------------------------
__global__ __launch_bounds__(128) void proj_kernel(
    const _Float16* __restrict__ Ctx, const _Float16* __restrict__ Wt,
    const float* __restrict__ bias, float* __restrict__ out) {
  __shared__ _Float16 As[2][16][32];
  int tid = threadIdx.x;
  int wave = tid >> 5, lane = tid & 31;
  int nl = lane & 15, hh = lane >> 4;
  int m0 = blockIdx.x * 16;
  int n0 = blockIdx.y * 64 + wave * 16;
  unsigned ldsA = (unsigned)(uintptr_t)&As[0][0][0];
  const _Float16* Xbase = Ctx + (size_t)m0 * DM;

  auto issueA = [&](int k0, int buf) {
    if (tid < 64) {
      int row = tid >> 2, part = tid & 3;
      unsigned long long g =
          (unsigned long long)(uintptr_t)(Xbase + (size_t)row * DM + k0) +
          (unsigned long long)part * 16u;
      asm volatile("global_load_async_to_lds_b128 %0, %1, off"
                   :: "v"(ldsA + (unsigned)buf * 1024u + (unsigned)tid * 16u),
                      "v"(g) : "memory");
    }
  };

  v8f acc = {};
  issueA(0, 0);
  for (int k0 = 0; k0 < DM; k0 += 32) {
    int cur = (k0 >> 5) & 1;
    if (k0 + 32 < DM) {
      issueA(k0 + 32, cur ^ 1);
      asm volatile("s_wait_asynccnt 0x1" ::: "memory");
    } else {
      asm volatile("s_wait_asynccnt 0x0" ::: "memory");
    }
    __syncthreads();
    v16h a, bfr;
#pragma unroll
    for (int e = 0; e < 16; ++e) a[e] = As[cur][nl][a_elem_k(e, hh)];
    const _Float16* wp = &Wt[(size_t)(n0 + nl) * DM + k0 + 16 * hh];
#pragma unroll
    for (int e = 0; e < 16; ++e) bfr[e] = wp[e];
    acc = __builtin_amdgcn_wmma_f32_16x16x32_f16(false, a, false, bfr,
                                                 (short)0, acc, false, false);
    __syncthreads();
  }
  float bb = bias[n0 + nl];
#pragma unroll
  for (int r = 0; r < 8; ++r)
    out[(size_t)(m0 + r + 8 * hh) * DM + n0 + nl] = acc[r] + bb;
}

// ---------------------------------------------------------------------------
extern "C" void kernel_launch(void* const* d_in, const int* in_sizes, int n_in,
                              void* d_out, int out_size, void* d_ws,
                              size_t ws_size, hipStream_t stream) {
  const float* hidden        = (const float*)d_in[0];
  const unsigned char* mask  = (const unsigned char*)d_in[1];
  const float* alibi         = (const float*)d_in[2];
  const float* W_qkv         = (const float*)d_in[3];
  const float* b_qkv         = (const float*)d_in[4];
  const float* W_proj        = (const float*)d_in[5];
  const float* b_proj        = (const float*)d_in[6];
  float* out = (float*)d_out;

  const size_t qkv_elems = (size_t)NB * NH * SQ * DHd;  // 4.19M
  const int    hid_elems = NB * SQ * DM;                // 4.19M
  char* ws = (char*)d_ws;
  _Float16* Wqt = (_Float16*)ws; ws += (size_t)1536 * 512 * sizeof(_Float16);
  _Float16* Wpt = (_Float16*)ws; ws += (size_t)512 * 512 * sizeof(_Float16);
  _Float16* Xh  = (_Float16*)ws; ws += (size_t)hid_elems * sizeof(_Float16);
  _Float16* Qh  = (_Float16*)ws; ws += qkv_elems * sizeof(_Float16);
  _Float16* Kh  = (_Float16*)ws; ws += qkv_elems * sizeof(_Float16);
  _Float16* Vt  = (_Float16*)ws; ws += qkv_elems * sizeof(_Float16);
  _Float16* Ctx = (_Float16*)ws;

  hcast_kernel<<<(hid_elems + 255) / 256, 256, 0, stream>>>(hidden, Xh, hid_elems);
  wcast_kernel<<<(1536 * 512 + 255) / 256, 256, 0, stream>>>(W_qkv, Wqt, 512, 1536);
  wcast_kernel<<<(512 * 512 + 255) / 256, 256, 0, stream>>>(W_proj, Wpt, 512, 512);
  qkv_kernel<<<dim3(NB * SQ / 16, 1536 / 64), 128, 0, stream>>>(
      Xh, Wqt, b_qkv, Qh, Kh, Vt);
  attn_kernel<<<NB * NH * SQ / 64, 128, 0, stream>>>(Qh, Kh, Vt, alibi, mask, Ctx);
  proj_kernel<<<dim3(NB * SQ / 16, DM / 64), 128, 0, stream>>>(Ctx, Wpt, b_proj, out);
}
